// YamRNNLayer_69939247448781
// MI455X (gfx1250) — matrix operand order repaired and checked by reading, not compiled, not run on hardware
//
#include <hip/hip_runtime.h>
#include <math.h>

#define Bq 64
#define Tq 512
#define Iq 1024
#define Hq 1024

typedef __bf16 bf16;
typedef __attribute__((ext_vector_type(16))) __bf16 v16bf;
typedef __attribute__((ext_vector_type(8)))  __bf16 v8bf;
typedef __attribute__((ext_vector_type(8)))  float  v8f;
typedef __attribute__((ext_vector_type(4)))  float  v4f;
typedef __attribute__((ext_vector_type(4)))  unsigned v4u;

// Pack two fp32 into two bf16 (round-half-up) in one dword:
// 2x v_add + 1x v_perm_b32.  out = {bf16(f1), bf16(f0)}.
__device__ __forceinline__ unsigned pk2bf(float f0, float f1) {
  unsigned u0 = __builtin_bit_cast(unsigned, f0) + 0x8000u;
  unsigned u1 = __builtin_bit_cast(unsigned, f1) + 0x8000u;
  return __builtin_amdgcn_perm(u1, u0, 0x07060302u);  // [u0.hi16, u1.hi16]
}

__device__ __forceinline__ bf16 f2bf(float f) {
  unsigned u = (__builtin_bit_cast(unsigned, f) + 0x8000u) >> 16;
  unsigned short h = (unsigned short)u;
  return __builtin_bit_cast(bf16, h);
}

// 8 contiguous floats -> v8bf via 4 perms.
__device__ __forceinline__ v8bf cvt8(const float* __restrict__ p) {
  v4f a = *(const v4f*)p;
  v4f b = *(const v4f*)(p + 4);
  v4u o;
  o[0] = pk2bf(a[0], a[1]);
  o[1] = pk2bf(a[2], a[3]);
  o[2] = pk2bf(b[0], b[1]);
  o[3] = pk2bf(b[2], b[3]);
  return __builtin_bit_cast(v8bf, o);
}

__device__ __forceinline__ v16bf cat16(v8bf a, v8bf b) {
  return __builtin_shufflevector(a, b, 0,1,2,3,4,5,6,7,8,9,10,11,12,13,14,15);
}

// ---------------------------------------------------------------------------
// fp32 -> bf16 bulk convert (memory-bound, one pass). count % 8 == 0.
// ---------------------------------------------------------------------------
__global__ void __launch_bounds__(256)
yam_cvt_kernel(const float* __restrict__ src, bf16* __restrict__ dst, int count8) {
  int i = blockIdx.x * blockDim.x + threadIdx.x;
  if (i < count8) {
    *(v8bf*)(dst + (size_t)i * 8) = cvt8(src + (size_t)i * 8);
  }
}

// ---------------------------------------------------------------------------
// h state init (fp32 + bf16 copies) and barrier counter reset
// ---------------------------------------------------------------------------
__global__ void yam_init_kernel(const float* __restrict__ h0,
                                float* __restrict__ hfp,
                                bf16* __restrict__ hbf,
                                unsigned* __restrict__ cnt) {
  int i = blockIdx.x * blockDim.x + threadIdx.x;
  if (i < Bq * Hq) {
    float v = h0[i];
    hfp[i] = v;
    hbf[i] = f2bf(v);
  }
  if (i == 0) *cnt = 0u;
}

// ---------------------------------------------------------------------------
// Phase 1: xp[m,n] = sum_k x[m,k] * W_ih[n,k] + b_ih[n]  (bf16 operands)
// M = 32768, N = 1024, K = 1024. WG = 128 threads (4 waves), tile 128Mx64N;
// each wave: 32Mx64N (2 M-tiles x 4 N-tiles, 8 f32 accumulators).
// Inner loop is pure b128 loads + 8 WMMAs (no conversion VALU).
// ---------------------------------------------------------------------------
__global__ void __launch_bounds__(128)
yam_xp_gemm_kernel(const bf16* __restrict__ xbf,
                   const bf16* __restrict__ wbf,
                   const float* __restrict__ b_ih,
                   float* __restrict__ xp) {
  const int lane = threadIdx.x & 31;
  const int wave = threadIdx.x >> 5;
  const int hf   = lane >> 4;
  const int l15  = lane & 15;
  const int m0   = blockIdx.x * 128 + wave * 32;   // 2 M-tiles of 16
  const int n0   = blockIdx.y * 64;                // 4 N-tiles of 16

  v8f acc[2][4];
#pragma unroll
  for (int mtl = 0; mtl < 2; ++mtl)
#pragma unroll
    for (int nt = 0; nt < 4; ++nt)
#pragma unroll
      for (int r = 0; r < 8; ++r) acc[mtl][nt][r] = 0.0f;

  const bf16* arow0 = xbf + (size_t)(m0 + l15) * Iq;
  const bf16* arow1 = xbf + (size_t)(m0 + 16 + l15) * Iq;

  for (int k0 = 0; k0 < Iq; k0 += 32) {
    // A 16x32 bf16: lane half0 -> K {0..7,16..23}, half1 -> {8..15,24..31}
    v16bf a0 = cat16(*(const v8bf*)(arow0 + k0 + 8 * hf),
                     *(const v8bf*)(arow0 + k0 + 16 + 8 * hf));
    v16bf a1 = cat16(*(const v8bf*)(arow1 + k0 + 8 * hf),
                     *(const v8bf*)(arow1 + k0 + 16 + 8 * hf));
#pragma unroll
    for (int nt = 0; nt < 4; ++nt) {
      // B 32x16 bf16: lane half0 -> K {0..7,16..23}, half1 -> {8..15,24..31}
      const bf16* bcol = wbf + (size_t)(n0 + nt * 16 + l15) * Iq + k0 + 8 * hf;
      v16bf b = cat16(*(const v8bf*)bcol, *(const v8bf*)(bcol + 16));
      acc[0][nt] = __builtin_amdgcn_wmma_f32_16x16x32_bf16(
          false, a0, false, b, (short)0, acc[0][nt], false, false);
      acc[1][nt] = __builtin_amdgcn_wmma_f32_16x16x32_bf16(
          false, a1, false, b, (short)0, acc[1][nt], false, false);
    }
  }

#pragma unroll
  for (int nt = 0; nt < 4; ++nt) {
    const int n = n0 + nt * 16 + l15;
    const float bias = b_ih[n];
#pragma unroll
    for (int mtl = 0; mtl < 2; ++mtl)
#pragma unroll
      for (int r = 0; r < 8; ++r) {
        const int m = m0 + mtl * 16 + r + 8 * hf;   // D: VGPR r <-> M=r+8*half
        xp[(size_t)m * Hq + n] = acc[mtl][nt][r] + bias;
      }
  }
}

// ---------------------------------------------------------------------------
// Phase 2: persistent scan. 64 workgroups x 256 threads (8 waves); each WG
// owns a 16-column slice of H. LDS: 64 KB bf16 W_hh1/W_hh2 slices
// (B-tile-swizzled 16B chunks) + 4 KB exchange. Waves 0-3: W_hh1 GEMM +
// elementwise; waves 4-7: W_hh2 GEMM. Grid barrier per timestep.
// ---------------------------------------------------------------------------
__global__ void __launch_bounds__(256)
yam_scan_kernel(const float* __restrict__ xp,
                const float* __restrict__ W1, const float* __restrict__ bh1,
                const float* __restrict__ W2, const float* __restrict__ bh2,
                float* __restrict__ hfp,      // 2 * B*H fp32 ping-pong
                bf16*  __restrict__ hbf,      // 2 * B*H bf16 ping-pong
                unsigned* __restrict__ barrier_cnt,
                float* __restrict__ out,      // [B,T,H] then [B,H] tail (h_T)
                int nblocks) {
  extern __shared__ char smem[];
  const int tid  = threadIdx.x;
  const int lane = tid & 31;
  const int wave = tid >> 5;
  const int hf   = lane >> 4;
  const int l15  = lane & 15;
  const int mt   = wave & 3;       // batch tile (16 rows)
  const int mat  = wave >> 2;      // 0 -> W_hh1, 1 -> W_hh2
  const int n0   = blockIdx.x * 16;

  // ---- Preload both W slices into LDS as bf16, B-tile swizzled. ----
  // chunk(kb,c,ll) -> W[col=n0+(ll&15)][K = 32*kb + 16*(ll>>4) + 8*c .. +8)
  // at base + kb*1024 + c*512 + ll*16.
  for (int m = 0; m < 2; ++m) {
    const float* Wm = m ? W2 : W1;
    char* base = smem + m * 32768;
    for (int cid = tid; cid < 2048; cid += 256) {
      const int kb = cid >> 6;
      const int c  = (cid >> 5) & 1;
      const int ll = cid & 31;
      const int col = n0 + (ll & 15);
      const int K   = kb * 32 + 16 * (ll >> 4) + c * 8;
      v8bf v = cvt8(Wm + (size_t)col * Hq + K);
      *(v8bf*)(base + kb * 1024 + c * 512 + ll * 16) = v;
    }
  }
  __syncthreads();

  const int ncol = n0 + l15;
  const float bias1 = bh1[ncol];
  const float bias2 = bh2[ncol];
  char* xbuf = smem + 65536;                 // 4 KB exchange region
  const char* bbase = smem + mat * 32768;

  for (int t = 0; t < Tq; ++t) {
    const int rd = t & 1, wr = rd ^ 1;
    const bf16* arow = hbf + (size_t)rd * (Bq * Hq) + (size_t)(mt * 16 + l15) * Hq;

    v8f acc;
#pragma unroll
    for (int r = 0; r < 8; ++r) acc[r] = 0.0f;

#pragma unroll 4
    for (int kb = 0; kb < 32; ++kb) {
      v16bf a = cat16(*(const v8bf*)(arow + kb * 32 + 16 * hf),
                      *(const v8bf*)(arow + kb * 32 + 8 + 16 * hf));
      v16bf b = cat16(*(const v8bf*)(bbase + kb * 1024 + lane * 16),
                      *(const v8bf*)(bbase + kb * 1024 + 512 + lane * 16));
      acc = __builtin_amdgcn_wmma_f32_16x16x32_bf16(
          false, a, false, b, (short)0, acc, false, false);
    }

    // W_hh2 waves publish their pre-activation tile via LDS.
    if (mat == 1) {
      float* xb = (float*)(xbuf + mt * 1024);
#pragma unroll
      for (int r = 0; r < 8; ++r) xb[(hf * 8 + r) * 16 + l15] = acc[r];
    }
    __syncthreads();

    if (mat == 0) {
      const float* xb    = (const float*)(xbuf + mt * 1024);
      const float* hprev = hfp + (size_t)rd * (Bq * Hq);
      float* hnf = hfp + (size_t)wr * (Bq * Hq);
      bf16*  hnb = hbf + (size_t)wr * (Bq * Hq);
#pragma unroll
      for (int r = 0; r < 8; ++r) {
        const int b = mt * 16 + 8 * hf + r;
        const size_t oidx = ((size_t)b * Tq + t) * Hq + ncol;
        const float xv = xp[oidx];
        const float h  = hprev[b * Hq + ncol];
        const float g2 = xb[(hf * 8 + r) * 16 + l15];
        const float v1 = tanhf(xv + acc[r] + bias1);
        const float v2 = tanhf(xv + g2 + bias2);
        const float hn = (1.0f - h) * v1 + h * v2;
        out[oidx] = hn;
        hnf[b * Hq + ncol] = hn;
        hnb[b * Hq + ncol] = f2bf(hn);
        if (t == Tq - 1) out[(size_t)Bq * Tq * Hq + (size_t)b * Hq + ncol] = hn;
        if (t + 1 < Tq) __builtin_prefetch(&xp[oidx + Hq], 0, 1);
      }
    }

    // ---- grid-wide barrier (monotonic counter, sense-free) ----
    __threadfence();
    __syncthreads();
    if (tid == 0) {
      __hip_atomic_fetch_add(barrier_cnt, 1u, __ATOMIC_ACQ_REL,
                             __HIP_MEMORY_SCOPE_AGENT);
      const unsigned target = (unsigned)nblocks * (unsigned)(t + 1);
      while (__hip_atomic_load(barrier_cnt, __ATOMIC_ACQUIRE,
                               __HIP_MEMORY_SCOPE_AGENT) < target) {
        __builtin_amdgcn_s_sleep(1);
      }
    }
    __syncthreads();
  }
}

// ---------------------------------------------------------------------------
extern "C" void kernel_launch(void* const* d_in, const int* in_sizes, int n_in,
                              void* d_out, int out_size, void* d_ws, size_t ws_size,
                              hipStream_t stream) {
  (void)in_sizes; (void)n_in; (void)out_size; (void)ws_size;
  const float* x    = (const float*)d_in[0];
  const float* h0   = (const float*)d_in[1];
  const float* Wih  = (const float*)d_in[2];
  const float* bih  = (const float*)d_in[3];
  const float* Whh1 = (const float*)d_in[4];
  const float* bhh1 = (const float*)d_in[5];
  const float* Whh2 = (const float*)d_in[6];
  const float* bhh2 = (const float*)d_in[7];
  float* out = (float*)d_out;

  char* ws = (char*)d_ws;
  size_t off = 0;
  unsigned* cnt = (unsigned*)(ws + off);            off += 256;
  float* hfp = (float*)(ws + off);                  off += (size_t)2 * Bq * Hq * 4;
  bf16*  hbf = (bf16*)(ws + off);                   off += (size_t)2 * Bq * Hq * 2;
  float* xp  = (float*)(ws + off);                  off += (size_t)Bq * Tq * Hq * 4;
  bf16*  xbf = (bf16*)(ws + off);                   off += (size_t)Bq * Tq * Iq * 2;
  bf16*  wbf = (bf16*)(ws + off);                   off += (size_t)Hq * Iq * 2;

  yam_init_kernel<<<(Bq * Hq + 255) / 256, 256, 0, stream>>>(h0, hfp, hbf, cnt);

  // Bulk fp32->bf16 conversion of x and W_ih (one HBM pass each).
  const int cx = (Bq * Tq * Iq) / 8;
  yam_cvt_kernel<<<(cx + 255) / 256, 256, 0, stream>>>(x, xbf, cx);
  const int cw = (Hq * Iq) / 8;
  yam_cvt_kernel<<<(cw + 255) / 256, 256, 0, stream>>>(Wih, wbf, cw);

  dim3 g1(Bq * Tq / 128, Hq / 64);   // 256 x 16 workgroups, 128 threads each
  yam_xp_gemm_kernel<<<g1, 128, 0, stream>>>(xbf, wbf, bih, xp);

  const int nblocks = Hq / 16;       // 64 persistent workgroups
  yam_scan_kernel<<<nblocks, 256, 65536 + 4096, stream>>>(
      xp, Whh1, bhh1, Whh2, bhh2, hfp, hbf, cnt, out, nblocks);
}
